// SimCLRLoss_2508260900941
// MI455X (gfx1250) — compile-verified, matrix operand order checked by reading, not run on hardware
//
#include <hip/hip_runtime.h>
#include <hip/hip_bf16.h>
#include <math.h>

typedef __attribute__((ext_vector_type(16))) _Float16 v16h;
typedef __attribute__((ext_vector_type(8)))  _Float16 v8h;
typedef __attribute__((ext_vector_type(8)))  float    v8f;
typedef __attribute__((ext_vector_type(4)))  int      v4i;

#define D_DIM    512
#define TEMP_INV 2.0f     // 1 / 0.5
#define NEG_BIG  (-1.0e30f)
#define LSE_SHIFT 2.0f    // |cos|/temp <= 2, fixed logsumexp shift

// ---------------- async global->LDS staging (CDNA5 path) ----------------
#if defined(__AMDGCN__) && defined(__has_builtin)
#if __has_builtin(__builtin_amdgcn_global_load_async_to_lds_b128)
#define HAVE_ASYNC_LDS 1
#endif
#endif

__device__ __forceinline__ void async_copy16(const _Float16* g, _Float16* l)
{
#ifdef HAVE_ASYNC_LDS
    __builtin_amdgcn_global_load_async_to_lds_b128(
        (__attribute__((address_space(1))) v4i*)g,
        (__attribute__((address_space(3))) v4i*)l,
        0, 0);
#else
    *(v8h*)l = *(const v8h*)g;          // fallback: vmem load + ds_store
#endif
}

__device__ __forceinline__ void async_wait()
{
#ifdef HAVE_ASYNC_LDS
#if __has_builtin(__builtin_amdgcn_s_wait_asynccnt)
    __builtin_amdgcn_s_wait_asynccnt(0);
#else
    asm volatile("s_wait_asynccnt 0x0" ::: "memory");
#endif
#endif
}

// ------------------------------------------------------------------
// Kernel 1: per-row L2 normalize, f32 -> f16, concat [z_i; z_j].
// ------------------------------------------------------------------
__global__ void __launch_bounds__(256) nrm_kernel(const float* __restrict__ zi,
                                                  const float* __restrict__ zj,
                                                  _Float16* __restrict__ zn,
                                                  int Brows)
{
    const int lane = threadIdx.x & 31;
    const int wave = threadIdx.x >> 5;
    const int row  = blockIdx.x * 8 + wave;
    const float* src = (row < Brows) ? (zi + (size_t)row * D_DIM)
                                     : (zj + (size_t)(row - Brows) * D_DIM);
    float v[16];
    float ss = 0.f;
#pragma unroll
    for (int i = 0; i < 16; ++i) {
        v[i] = src[lane + i * 32];
        ss += v[i] * v[i];
    }
#pragma unroll
    for (int off = 16; off; off >>= 1)
        ss += __shfl_xor(ss, off, 32);
    const float scale = 1.0f / fmaxf(sqrtf(ss), 1e-8f);
    _Float16* dst = zn + (size_t)row * D_DIM;
#pragma unroll
    for (int i = 0; i < 16; ++i)
        dst[lane + i * 32] = (_Float16)(v[i] * scale);
}

// ------------------------------------------------------------------
// Fragment loaders: 16-bit A-style 16x32 fragment (ISA 7.12.2).
// Gram symmetry: B operand column n == row n of zn, same layout.
// ------------------------------------------------------------------
__device__ __forceinline__ v16h pack_frag(v8h lo, v8h hh)
{
    v16h r;
#pragma unroll
    for (int i = 0; i < 8; ++i) { r[i] = lo[i]; r[i + 8] = hh[i]; }
    return r;
}

__device__ __forceinline__ v16h load_frag_g(const _Float16* __restrict__ base,
                                            int row0, int k0, int lane)
{
    const int m  = lane & 15;
    const int hi = lane >> 4;
    const _Float16* p = base + (size_t)(row0 + m) * D_DIM + (k0 + hi * 8);
    return pack_frag(*(const v8h*)p, *(const v8h*)(p + 16));
}

__device__ __forceinline__ v16h load_frag_lds(const _Float16* buf, int k0, int lane)
{
    const int m  = lane & 15;
    const int hi = lane >> 4;
    const _Float16* p = buf + m * D_DIM + (k0 + hi * 8);
    return pack_frag(*(const v8h*)p, *(const v8h*)(p + 16));
}

// Partial-logsumexp epilogue for one 16x16 accumulator tile.
__device__ __forceinline__ void lse_epilogue(const v8f& acc, int ct, int rt, int ctp,
                                             float* s, float* p, int n, int hi)
{
    const bool diag_tile = (ct == rt);
    const bool pos_tile  = (ct == ctp);
#pragma unroll
    for (int v = 0; v < 8; ++v) {
        const int  m   = v + (hi << 3);
        const bool onm = (n == m);             // i%16 == j%16 slot
        float x = acc[v] * TEMP_INV;           // bounded in [-2, 2]
        p[v] = (pos_tile && onm) ? x : p[v];
        float e = __expf(x - LSE_SHIFT);
        s[v] += (diag_tile && onm) ? 0.0f : e;
    }
}

// ------------------------------------------------------------------
// Kernel 2: fused Gram (WMMA f16->f32) + partial fixed-shift LSE.
// Grid: (Nt/16 row groups) x (2 column halves). Block = 8 waves;
// each wave owns TWO 16-row tiles (rt0, rt0+8): two independent
// WMMAs share every LDS B fragment -> half the LDS bytes per WMMA
// and two interleaved accumulator chains. Column tiles are staged
// to LDS once per block, double-buffered via async global->LDS.
// ------------------------------------------------------------------
__global__ void __launch_bounds__(256) gram_lse_kernel(const _Float16* __restrict__ zn,
                                                       float* __restrict__ sbuf,
                                                       float* __restrict__ pbuf,
                                                       int Nt, int Bt)
{
    __shared__ _Float16 lds_buf[2][16 * D_DIM];   // 2 x 16KB column tiles

    const int tid  = threadIdx.x;
    const int lane = tid & 31;
    const int wave = tid >> 5;
    const int rg   = blockIdx.x >> 1;             // row group of 16 tiles
    const int half = blockIdx.x & 1;              // column half
    const int ctBeg = half * (Nt / 2);
    const int ctEnd = ctBeg + (Nt / 2);
    const int rt0  = rg * 16 + wave;
    const int rt1  = rt0 + 8;
    const int n    = lane & 15;
    const int hi   = lane >> 4;
    const int ctp0 = (rt0 + Bt) % Nt;
    const int ctp1 = (rt1 + Bt) % Nt;
    const int Nrows = Nt * 16;

    // A fragments for this wave's 32 rows (two tiles), all K=512
    v16h a0[16], a1[16];
#pragma unroll
    for (int kk = 0; kk < 16; ++kk) {
        a0[kk] = load_frag_g(zn, rt0 * 16, kk * 32, lane);
        a1[kk] = load_frag_g(zn, rt1 * 16, kk * 32, lane);
    }

    float s0[8], p0[8], s1[8], p1[8];
#pragma unroll
    for (int v = 0; v < 8; ++v) {
        s0[v] = 0.f; p0[v] = NEG_BIG;
        s1[v] = 0.f; p1[v] = NEG_BIG;
    }

    // stage first column tile (16KB contiguous; 4 x 16B per thread)
    {
        const _Float16* src = zn + (size_t)ctBeg * 16 * D_DIM;
#pragma unroll
        for (int i = 0; i < 4; ++i) {
            const int off = (tid + 256 * i) * 8;  // 8 halfs = 16 bytes
            async_copy16(src + off, &lds_buf[ctBeg & 1][off]);
        }
    }

    for (int ct = ctBeg; ct < ctEnd; ++ct) {
        const int cur = ct & 1;
        async_wait();
        __syncthreads();                          // staged tile visible to all

        if (ct + 1 < ctEnd) {                     // prefetch next tile
            const _Float16* src = zn + (size_t)(ct + 1) * 16 * D_DIM;
#pragma unroll
            for (int i = 0; i < 4; ++i) {
                const int off = (tid + 256 * i) * 8;
                async_copy16(src + off, &lds_buf[cur ^ 1][off]);
            }
        }

        v8f acc0 = {}, acc1 = {};
#pragma unroll
        for (int kk = 0; kk < 16; ++kk) {
            v16h b = load_frag_lds(&lds_buf[cur][0], kk * 32, lane);
            acc0 = __builtin_amdgcn_wmma_f32_16x16x32_f16(
                       false, a0[kk], false, b, (short)0, acc0, false, false);
            acc1 = __builtin_amdgcn_wmma_f32_16x16x32_f16(
                       false, a1[kk], false, b, (short)0, acc1, false, false);
        }

        lse_epilogue(acc0, ct, rt0, ctp0, s0, p0, n, hi);
        lse_epilogue(acc1, ct, rt1, ctp1, s1, p1, n, hi);
        __syncthreads();                          // all done reading lds_buf[cur]
    }

    // merge the 16 column slots (butterfly over lanes sharing a row set)
#pragma unroll
    for (int v = 0; v < 8; ++v) {
#pragma unroll
        for (int off = 8; off; off >>= 1) {
            s0[v] += __shfl_xor(s0[v], off, 16);
            s1[v] += __shfl_xor(s1[v], off, 16);
            p0[v]  = fmaxf(p0[v], __shfl_xor(p0[v], off, 16));
            p1[v]  = fmaxf(p1[v], __shfl_xor(p1[v], off, 16));
        }
    }

    if ((lane & 15) == 0) {
#pragma unroll
        for (int v = 0; v < 8; ++v) {
            const int m = v + (hi << 3);
            sbuf[half * Nrows + rt0 * 16 + m] = s0[v];
            pbuf[half * Nrows + rt0 * 16 + m] = p0[v];
            sbuf[half * Nrows + rt1 * 16 + m] = s1[v];
            pbuf[half * Nrows + rt1 * 16 + m] = p1[v];
        }
    }
}

// ------------------------------------------------------------------
// Kernel 3: merge column halves + deterministic tree reduction.
// ------------------------------------------------------------------
__global__ void __launch_bounds__(256) reduce_kernel(const float* __restrict__ sbuf,
                                                     const float* __restrict__ pbuf,
                                                     float* __restrict__ out, int N)
{
    __shared__ float sm[256];
    float acc = 0.f;
    for (int i = threadIdx.x; i < N; i += 256) {
        float s = sbuf[i] + sbuf[N + i];
        float p = fmaxf(pbuf[i], pbuf[N + i]);
        acc += (LSE_SHIFT + logf(s)) - p;
    }
    sm[threadIdx.x] = acc;
    __syncthreads();
    for (int st = 128; st; st >>= 1) {
        if ((int)threadIdx.x < st) sm[threadIdx.x] += sm[threadIdx.x + st];
        __syncthreads();
    }
    if (threadIdx.x == 0) out[0] = sm[0] / (float)N;
}

extern "C" void kernel_launch(void* const* d_in, const int* in_sizes, int n_in,
                              void* d_out, int out_size, void* d_ws, size_t ws_size,
                              hipStream_t stream)
{
    (void)n_in; (void)out_size; (void)ws_size;
    const float* zi = (const float*)d_in[0];
    const float* zj = (const float*)d_in[1];
    const int Brows = in_sizes[0] / D_DIM;   // 4096
    const int N     = 2 * Brows;             // 8192
    const int Nt    = N / 16;                // 512 row/column tiles
    const int Bt    = Brows / 16;            // 256 tile offset of positives

    _Float16* zn   = (_Float16*)d_ws;                                       // 8 MB
    float*    sbuf = (float*)((char*)d_ws + (size_t)N * D_DIM * sizeof(_Float16));
    float*    pbuf = sbuf + 2 * (size_t)N;

    nrm_kernel<<<N / 8, 256, 0, stream>>>(zi, zj, zn, Brows);
    gram_lse_kernel<<<(Nt / 16) * 2, 256, 0, stream>>>(zn, sbuf, pbuf, Nt, Bt);
    reduce_kernel<<<1, 256, 0, stream>>>(sbuf, pbuf, (float*)d_out, N);
}
